// convlstm_unit_remnet_fb_68779606278793
// MI455X (gfx1250) — matrix operand order repaired and checked by reading, not compiled, not run on hardware
//
#include <hip/hip_runtime.h>

typedef __attribute__((ext_vector_type(16))) _Float16 v16h;
typedef __attribute__((ext_vector_type(8)))  _Float16 v8h;
typedef __attribute__((ext_vector_type(8)))  float    v8f;

#define NB   64
#define HC   64
#define MG   48
#define REC  5
#define PIX  4096      // 64*64
#define HPIX 1024      // 32*32

__device__ __forceinline__ v16h cat16(v8h lo, v8h hi) {
  return __builtin_shufflevector(lo, hi, 0,1,2,3,4,5,6,7,8,9,10,11,12,13,14,15);
}
__device__ __forceinline__ float sigm(float x) { return 1.0f / (1.0f + __expf(-x)); }
__device__ __forceinline__ float lrelu(float x) { return x >= 0.f ? x : 0.2f * x; }

// CDNA5 async DMA: global -> LDS, tracked by ASYNCcnt (ISA 15.18.3 op 98).
__device__ __forceinline__ void async_g2l_b128(unsigned lds_off, const void* gaddr) {
  asm volatile("global_load_async_to_lds_b128 %0, %1, off"
               :: "v"(lds_off), "v"(gaddr) : "memory");
}
__device__ __forceinline__ void wait_async0() {
  asm volatile("s_wait_asynccnt 0x0" ::: "memory");
}

// ---------------------------------------------------------------------------
// K0: convert weights f32 -> f16, pre-swizzled into WMMA B-fragment layout.
// B-frag element j of lane l holds k = (j&7) + 16*(j>=8) + 8*(l>=16), column n = lane%16.
// ---------------------------------------------------------------------------
__global__ void k0_weights(const float* __restrict__ conv_w,
                           const float* __restrict__ convfuse_w,
                           const float* __restrict__ convssm_w,
                           _Float16* __restrict__ w_frag,
                           _Float16* __restrict__ wfuse_frag,
                           _Float16* __restrict__ wssm_frag) {
  int idx = blockIdx.x * 256 + threadIdx.x;
  if (idx < 368640) {                       // conv3x3: 36*20*32*16
    int j = idx & 15, lane = (idx >> 4) & 31, t = idx >> 9;
    int kt = t / 20, nt = t % 20;
    int k = kt * 32 + (j & 7) + ((j >= 8) ? 16 : 0) + ((lane >= 16) ? 8 : 0);
    int rs = k >> 7, ic = k & 127;
    int n = nt * 16 + (lane & 15);
    w_frag[idx] = (_Float16)conv_w[(n * 128 + ic) * 9 + rs];
    return;
  }
  idx -= 368640;
  if (idx < 8192) {                         // fuse: 4*4*32*16
    int j = idx & 15, lane = (idx >> 4) & 31, t = idx >> 9;
    int kt = t / 4, nt = t % 4;
    int k = kt * 32 + (j & 7) + ((j >= 8) ? 16 : 0) + ((lane >= 16) ? 8 : 0);
    int n = nt * 16 + (lane & 15);
    wfuse_frag[idx] = (_Float16)((k < 112) ? convfuse_w[n * 112 + k] : 0.0f);
    return;
  }
  idx -= 8192;
  if (idx < 49152) {                        // ssm: 4*8*3*32*16
    int j = idx & 15, lane = (idx >> 4) & 31, t = idx >> 9;
    int nt = t % 3, kt = (t / 3) & 7, p = t / 24;
    int py = p >> 1, px = p & 1;
    int k = kt * 32 + (j & 7) + ((j >= 8) ? 16 : 0) + ((lane >= 16) ? 8 : 0);
    int tap = k >> 6, ic = k & 63;
    int tyy = tap >> 1, txx = tap & 1;
    int ky = (1 - py) + 2 * tyy, kx = (1 - px) + 2 * txx;
    int n = nt * 16 + (lane & 15);
    float v = (ic < MG) ? convssm_w[((ic * MG + n) * 4 + ky) * 4 + kx] : 0.0f;
    wssm_frag[idx] = (_Float16)v;
  }
}

// ---------------------------------------------------------------------------
// KX: build channel-last f16 copy of concat(x,h): xh_cl[b][pix][128].
// ---------------------------------------------------------------------------
__global__ __launch_bounds__(256) void kx_xh(const float* __restrict__ x,
                                             const float* __restrict__ h,
                                             _Float16* __restrict__ xh_cl) {
  __shared__ __align__(16) _Float16 buf[128 * 136];   // padded rows: 34816 B
  int b = blockIdx.x >> 5, chunk = blockIdx.x & 31;
  int pix0 = chunk * 128;
  int t = threadIdx.x;
  int tpix = t & 127, thalf = t >> 7;
  const float* xb = x + (long)b * 64 * PIX;
  const float* hb = h + (long)b * HC * PIX;
  for (int ic0 = 0; ic0 < 128; ic0 += 2) {
    int ic = ic0 + thalf;
    float v = (ic < 64) ? xb[ic * PIX + pix0 + tpix] : hb[(ic - 64) * PIX + pix0 + tpix];
    buf[tpix * 136 + ic] = (_Float16)v;
  }
  __syncthreads();
  _Float16* dst = xh_cl + ((long)b * PIX + pix0) * 128;
  for (int u = t; u < 2048; u += 256) {               // 128 pix * 16 parts
    int p = u >> 4, part = u & 15;
    *(v8h*)(dst + p * 128 + part * 8) = *(const v8h*)(buf + p * 136 + part * 8);
  }
}

// ---------------------------------------------------------------------------
__global__ void kzero(float* __restrict__ q_avg) {
  int i = blockIdx.x * 256 + threadIdx.x;
  if (i < NB * MG) q_avg[i] = 0.f;
}

// ---------------------------------------------------------------------------
// K1a: rm_avg[b][r][m] = mean over 32x32.  One block per (b,r,m).
// ---------------------------------------------------------------------------
__global__ void k1a_rmavg(const float* __restrict__ rm, float* __restrict__ rm_avg) {
  int bid = blockIdx.x;                     // (b*5+r)*48+m
  const float4* src = (const float4*)(rm + (long)bid * HPIX);
  int t = threadIdx.x;
  float4 v = src[t];
  float s = v.x + v.y + v.z + v.w;
  __shared__ float red[256];
  red[t] = s; __syncthreads();
  for (int o = 128; o > 0; o >>= 1) { if (t < o) red[t] += red[t + o]; __syncthreads(); }
  if (t == 0) rm_avg[bid] = red[0] * (1.0f / 1024.0f);
}

// ---------------------------------------------------------------------------
// K1b: q_avg[b][m] += mean over pixels of leaky(convc(c)).  Grid: b*16 chunks.
// ---------------------------------------------------------------------------
__global__ __launch_bounds__(256) void k1b_qavg(const float* __restrict__ c,
                                                const float* __restrict__ wc,
                                                const float* __restrict__ bc,
                                                float* __restrict__ q_avg) {
  __shared__ float w_s[MG * HC];
  __shared__ float b_s[MG];
  int t = threadIdx.x;
  for (int i = t; i < MG * HC; i += 256) w_s[i] = wc[i];
  if (t < MG) b_s[t] = bc[t];
  __syncthreads();
  int b = blockIdx.x >> 4, chunk = blockIdx.x & 15;
  int p = chunk * 256 + t;
  const float* cb = c + (long)b * HC * PIX + p;
  float acc[MG];
#pragma unroll
  for (int m = 0; m < MG; m++) acc[m] = b_s[m];
  for (int ic = 0; ic < HC; ic++) {
    float cv = cb[ic * PIX];
#pragma unroll
    for (int m = 0; m < MG; m++) acc[m] = fmaf(w_s[m * HC + ic], cv, acc[m]);
  }
  for (int m = 0; m < MG; m++) {
    float v = lrelu(acc[m]) * (1.0f / (float)PIX);
    for (int off = 16; off > 0; off >>= 1) v += __shfl_xor(v, off, 32);
    if ((t & 31) == 0) atomicAdd(&q_avg[b * MG + m], v);
  }
}

// ---------------------------------------------------------------------------
// K1c: cosine sim + softmax -> attn[b][r].
// ---------------------------------------------------------------------------
__global__ void k1c_attn(const float* __restrict__ q_avg,
                         const float* __restrict__ rm_avg,
                         float* __restrict__ attn) {
  int b = blockIdx.x, l = threadIdx.x;
  const float* q = q_avg + b * MG;
  float qn2 = 0.f;
  for (int m = 0; m < MG; m++) qn2 += q[m] * q[m];
  float qn = fmaxf(sqrtf(qn2), 1e-8f);
  float cosr = 0.f;
  if (l < REC) {
    const float* rv = rm_avg + (b * REC + l) * MG;
    float dot = 0.f, mn2 = 0.f;
    for (int m = 0; m < MG; m++) { dot += q[m] * rv[m]; mn2 += rv[m] * rv[m]; }
    float mn = fmaxf(sqrtf(mn2), 1e-8f);
    cosr = dot / (qn * mn);
  }
  float mx = -1e30f;
  for (int r = 0; r < REC; r++) mx = fmaxf(mx, __shfl(cosr, r, 32));
  float ex = (l < REC) ? __expf(cosr - mx) : 0.f;
  float sum = 0.f;
  for (int r = 0; r < REC; r++) sum += __shfl(ex, r, 32);
  if (l < REC) attn[b * REC + l] = ex / sum;
}

// ---------------------------------------------------------------------------
// K2a: mt_cl[b][pix][ch(64 pad)] = sum_r attn*rm  (channel-last f16, padded).
// ---------------------------------------------------------------------------
__global__ void k2a_mt(const float* __restrict__ rm, const float* __restrict__ attn,
                       _Float16* __restrict__ mt_cl) {
  int b = blockIdx.x >> 6, ch = blockIdx.x & 63;
  int t = threadIdx.x;
  float a0 = attn[b * REC + 0], a1 = attn[b * REC + 1], a2 = attn[b * REC + 2];
  float a3 = attn[b * REC + 3], a4 = attn[b * REC + 4];
  for (int i = 0; i < 4; i++) {
    int pix = t + i * 256;
    float v = 0.f;
    if (ch < MG) {
      long base = ((long)(b * REC) * MG + ch) * HPIX + pix;
      const long st = (long)MG * HPIX;
      v = a0 * rm[base] + a1 * rm[base + st] + a2 * rm[base + 2 * st]
        + a3 * rm[base + 3 * st] + a4 * rm[base + 4 * st];
    }
    mt_cl[((long)b * HPIX + pix) * 64 + ch] = (_Float16)v;
  }
}

// ---------------------------------------------------------------------------
// K2b: ConvTranspose2d 4x4/s2/p1 (48->48) as 4 parity-class 2x2 convs, WMMA.
// ---------------------------------------------------------------------------
__global__ __launch_bounds__(256) void k2b_ssm(const _Float16* __restrict__ mt_cl,
                                               const _Float16* __restrict__ wssm_frag,
                                               const float* __restrict__ bssm,
                                               _Float16* __restrict__ mtup) {
  __shared__ __align__(16) _Float16 patch[9 * 9 * 64];   // 10368 B
  __shared__ __align__(16) _Float16 obuf[64 * 64];       // 8192 B
  int bid = blockIdx.x;
  int tile = bid & 15, p = (bid >> 4) & 3, b = bid >> 6;
  int py = p >> 1, px = p & 1;
  int ty0 = (tile >> 2) * 8, tx0 = (tile & 3) * 8;
  int iys = ty0 + py - 1, ixs = tx0 + px - 1;
  int t = threadIdx.x;
  unsigned lds_base = (unsigned)(unsigned long long)(void*)patch;
  for (int u = t; u < 81 * 8; u += 256) {
    int pos = u >> 3, part = u & 7;
    int dy = pos / 9, dx = pos % 9;
    int iy = iys + dy, ix = ixs + dx;
    unsigned dst = (unsigned)(((dy * 9 + dx) * 64 + part * 8) * 2);
    if (iy >= 0 && iy < 32 && ix >= 0 && ix < 32) {
      const _Float16* src = mt_cl + (((long)b * HPIX + iy * 32 + ix) * 64 + part * 8);
      async_g2l_b128(lds_base + dst, (const void*)src);
    } else {
      v8h z = {};
      *(v8h*)(patch + (dy * 9 + dx) * 64 + part * 8) = z;
    }
  }
  for (int u = t; u < 512; u += 256) {      // zero obuf (covers pad chs 48..63)
    v8h z = {};
    *(v8h*)(obuf + u * 8) = z;
  }
  wait_async0();
  __syncthreads();
  int wv = t >> 5, lane = t & 31;
  int lhalf = (lane >= 16) ? 8 : 0, lm = lane & 15;
  v8f zero = {};
  for (int tt = wv; tt < 12; tt += 8) {
    int mtile = tt & 3, ntile = tt >> 2;
    int m = mtile * 16 + lm;
    int ly = m >> 3, lx = m & 7;
    v16h af[8], bf[8];
#pragma unroll
    for (int kt = 0; kt < 8; kt++) {
      int tap = kt >> 1, icb = (kt & 1) * 32;
      int tyy = tap >> 1, txx = tap & 1;
      int base = ((ly + 1 - tyy) * 9 + (lx + 1 - txx)) * 64 + icb + lhalf;
      af[kt] = cat16(*(const v8h*)(patch + base), *(const v8h*)(patch + base + 16));
      unsigned widx = (((unsigned)(p * 8 + kt) * 3u + (unsigned)ntile) * 32u
                       + (unsigned)lane) * 16u;
      bf[kt] = cat16(*(const v8h*)(wssm_frag + widx),
                     *(const v8h*)(wssm_frag + widx + 8));
    }
    v8f a4[4] = {zero, zero, zero, zero};
#pragma unroll
    for (int kt = 0; kt < 8; kt++)
      a4[kt & 3] = __builtin_amdgcn_wmma_f32_16x16x32_f16(false, af[kt], false, bf[kt],
                                                          (short)0, a4[kt & 3],
                                                          false, false);
    v8f acc = (a4[0] + a4[1]) + (a4[2] + a4[3]);
    int n = ntile * 16 + lm;
    float bias = bssm[n];
#pragma unroll
    for (int v = 0; v < 8; v++) {
      int mm = mtile * 16 + v + lhalf;
      obuf[mm * 64 + n] = (_Float16)lrelu(acc[v] + bias);
    }
  }
  __syncthreads();
  for (int u = t; u < 512; u += 256) {      // coalesced b128 writeout
    int pp = u >> 3, part = u & 7;
    int lly = pp >> 3, llx = pp & 7;
    int oy = 2 * (ty0 + lly) + py, ox = 2 * (tx0 + llx) + px;
    *(v8h*)(mtup + (((long)b * 64 + oy) * 64 + ox) * 64 + part * 8) =
        *(const v8h*)(obuf + pp * 64 + part * 8);
  }
}

// ---------------------------------------------------------------------------
// K3: fused main kernel, DUAL-TILE: 128 pixels (8x16 region) per block.
// 320 threads = 10 waves; wave = one N-pair x ALL 8 M-tiles (16 accumulators),
// so each weight fragment is read exactly once per block and the block count
// is halved (halves weight L2 traffic).  Phase A: async DMA from xh_cl.
// Epilogue runs per 64-pixel half, reusing the same LDS buffers.
// ---------------------------------------------------------------------------
__global__ __launch_bounds__(320) void k3_main(const _Float16* __restrict__ xh_cl,
                                               const float* __restrict__ h,
                                               const float* __restrict__ c,
                                               const _Float16* __restrict__ w_frag,
                                               const float* __restrict__ conv_b,
                                               const _Float16* __restrict__ wfuse_frag,
                                               const float* __restrict__ fuse_b,
                                               const _Float16* __restrict__ mtup,
                                               float* __restrict__ h_out,
                                               float* __restrict__ c_out) {
  __shared__ __align__(16) char smem[57344];
  _Float16* patch  = (_Float16*)smem;                 // phase A: [180][128] = 46080 B
  _Float16* gatesF = (_Float16*)smem;                 // epi:    [5][64][64] = 40960 B
  _Float16* A2     = (_Float16*)(smem + 40960);       // epi:    [64][128]  = 16384 B
  float*    hbuf   = (float*)smem;                    // epi E3: [64][64] f32 (dead i-gate)

  int bid = blockIdx.x;
  int tile = bid & 31, b = bid >> 5;                  // 8 y-tiles x 4 x-tiles
  int y0 = (tile >> 2) * 8, x0 = (tile & 3) * 16;
  int t = threadIdx.x;

  // ---- Phase A: async-DMA the f16 input patch 10x18x128 (halo zero-filled) ----
  const _Float16* xhb = xh_cl + (long)b * PIX * 128;
  unsigned patch_off = (unsigned)(unsigned long long)(void*)patch;
  for (int u = t; u < 2880; u += 320) {               // 180 pos * 16 b128 parts
    int pos = u >> 4, part = u & 15;
    int dy = pos / 18, dx = pos % 18;
    int gy = y0 + dy - 1, gx = x0 + dx - 1;
    if (gy >= 0 && gy < 64 && gx >= 0 && gx < 64) {
      async_g2l_b128(patch_off + (unsigned)((pos * 128 + part * 8) * 2),
                     xhb + (gy * 64 + gx) * 128 + part * 8);
    } else {
      v8h z = {};
      *(v8h*)(patch + pos * 128 + part * 8) = z;
    }
  }
  wait_async0();
  __syncthreads();

  // ---- GEMM: M=128 pixels, N=320, K=1152 ----
  int wv = t >> 5, lane = t & 31;
  int lhalf = (lane >= 16) ? 8 : 0, lm = lane & 15;
  int ntb = wv * 2;                                   // this wave's ntile base (0..18)
  int pbase[8];
#pragma unroll
  for (int mt = 0; mt < 8; mt++) {
    int th = mt >> 2, sub = mt & 3;
    int ml = sub * 16 + lm;
    pbase[mt] = (ml >> 3) * 18 + (ml & 7) + th * 8;   // row*18 + col
  }
  v8f zero = {};
  v8f acc[16];
#pragma unroll
  for (int i = 0; i < 16; i++) acc[i] = zero;

  auto loadB = [&](int kt, v16h* b2) {
    unsigned widx = (((unsigned)kt * 20u + (unsigned)ntb) * 32u + (unsigned)lane) * 16u;
#pragma unroll
    for (int q = 0; q < 2; q++)
      b2[q] = cat16(*(const v8h*)(w_frag + widx + (unsigned)q * 512u),
                    *(const v8h*)(w_frag + widx + (unsigned)q * 512u + 8u));
  };
  auto gemmStep = [&](int kt, v16h* b2) {
    int rs = kt >> 2, icb = (kt & 3) * 32;
    int r = rs / 3, s = rs % 3;
    int off = (r * 18 + s) * 128 + icb + lhalf;
#pragma unroll
    for (int half = 0; half < 2; half++) {
      v16h aT[4];
#pragma unroll
      for (int sub = 0; sub < 4; sub++) {
        int base = pbase[half * 4 + sub] * 128 + off;
        aT[sub] = cat16(*(const v8h*)(patch + base), *(const v8h*)(patch + base + 16));
      }
#pragma unroll
      for (int sub = 0; sub < 4; sub++)
#pragma unroll
        for (int q = 0; q < 2; q++)
          acc[(half * 4 + sub) * 2 + q] = __builtin_amdgcn_wmma_f32_16x16x32_f16(
              false, aT[sub], false, b2[q], (short)0, acc[(half * 4 + sub) * 2 + q],
              false, false);
    }
  };

  v16h bA[2], bB[2];
  loadB(0, bA);
  for (int kt = 0; kt < 36; kt += 2) {
    loadB(kt + 1, bB);                                // prefetch kt+1
    gemmStep(kt, bA);
    int kn = (kt + 2 < 36) ? kt + 2 : 0;              // prefetch kt+2 (clamped)
    loadB(kn, bA);
    gemmStep(kt + 1, bB);
  }

  // ---- Epilogue: one 64-pixel half at a time, reusing LDS buffers ----
  for (int th = 0; th < 2; th++) {
    int x0h = x0 + th * 8;
    __syncthreads();                                  // patch / previous-half LDS dead

    // E1: bias + nonlinearity -> gatesF
#pragma unroll
    for (int sub = 0; sub < 4; sub++)
#pragma unroll
      for (int q = 0; q < 2; q++) {
        int n = (ntb + q) * 16 + lm;
        int grp = n >> 6, nloc = n & 63;              // i,f,g,o,lam
        float bias = conv_b[n];
        v8f a = acc[(th * 4 + sub) * 2 + q];
#pragma unroll
        for (int v = 0; v < 8; v++) {
          int mm = sub * 16 + v + lhalf;
          float val = a[v] + bias;
          val = (grp == 2) ? tanhf(val) : sigm(val);
          gatesF[(grp * 64 + mm) * 64 + nloc] = (_Float16)val;
        }
      }
    __syncthreads();

    // E2: c_new = f*c + i*g ; build fuse A-tile [pixel][128]
    const float* cbp = c + (long)b * HC * PIX;
    float* cno = c_out + (long)b * HC * PIX;
    const _Float16* mtb = mtup + (long)b * PIX * 64;
    unsigned a2_off = (unsigned)(unsigned long long)(void*)A2;
    for (int e = t; e < 512; e += 320) {              // mt half of A2 via async DMA
      int pp = e >> 3, part = e & 7;
      int gy = y0 + (pp >> 3), gx = x0h + (pp & 7);
      async_g2l_b128(a2_off + (unsigned)((pp * 128 + 64 + part * 8) * 2),
                     mtb + (gy * 64 + gx) * 64 + part * 8);
    }
    for (int e = t; e < 4096; e += 320) {
      int ch = e >> 6, pp = e & 63;
      int gy = y0 + (pp >> 3), gx = x0h + (pp & 7);
      float iv = (float)gatesF[(0 * 64 + pp) * 64 + ch];
      float fv = (float)gatesF[(1 * 64 + pp) * 64 + ch];
      float gv = (float)gatesF[(2 * 64 + pp) * 64 + ch];
      float cn = fv * cbp[ch * PIX + gy * 64 + gx] + iv * gv;
      cno[ch * PIX + gy * 64 + gx] = cn;
      A2[pp * 128 + ch] = (_Float16)cn;
    }
    wait_async0();
    __syncthreads();

    // E3: fuse conv (M=64, N=64, K=128), two interleaved chains -> hbuf
    if (wv < 8) {
      int fm = wv & 3, ntbase = (wv >> 2) * 2;
      int mA = fm * 16 + lm;
      v8f fac0 = zero, fac1 = zero;
#pragma unroll
      for (int kt2 = 0; kt2 < 4; kt2++) {
        int base = mA * 128 + kt2 * 32 + lhalf;
        v16h a = cat16(*(const v8h*)(A2 + base), *(const v8h*)(A2 + base + 16));
        unsigned w0 = ((unsigned)(kt2 * 4 + ntbase) * 32u + (unsigned)lane) * 16u;
        unsigned w1 = ((unsigned)(kt2 * 4 + ntbase + 1) * 32u + (unsigned)lane) * 16u;
        v16h b0 = cat16(*(const v8h*)(wfuse_frag + w0),
                        *(const v8h*)(wfuse_frag + w0 + 8));
        v16h b1 = cat16(*(const v8h*)(wfuse_frag + w1),
                        *(const v8h*)(wfuse_frag + w1 + 8));
        fac0 = __builtin_amdgcn_wmma_f32_16x16x32_f16(false, a, false, b0, (short)0,
                                                      fac0, false, false);
        fac1 = __builtin_amdgcn_wmma_f32_16x16x32_f16(false, a, false, b1, (short)0,
                                                      fac1, false, false);
      }
#pragma unroll
      for (int qq = 0; qq < 2; qq++) {
        v8f fac = qq ? fac1 : fac0;
        int n = (ntbase + qq) * 16 + lm;
        float fb = fuse_b[n];
#pragma unroll
        for (int v = 0; v < 8; v++) {
          int mm = fm * 16 + v + lhalf;
          float hn = tanhf(fac[v] + fb);
          float ov = (float)gatesF[(3 * 64 + mm) * 64 + n];
          hbuf[mm * 64 + n] = ov * hn;                // h_new staged in LDS
        }
      }
    }
    __syncthreads();

    // E4: coalesced lambda-blend + h_out writeout
    const float* hb2 = h + (long)b * HC * PIX;
    float* hout = h_out + (long)b * HC * PIX;
    for (int e = t; e < 4096; e += 320) {
      int ch = e >> 6, pp = e & 63;
      int gy = y0 + (pp >> 3), gx = x0h + (pp & 7);
      float lv = (float)gatesF[(4 * 64 + pp) * 64 + ch];
      float hi = hb2[ch * PIX + gy * 64 + gx];
      hout[ch * PIX + gy * 64 + gx] = lv * hbuf[pp * 64 + ch] + (1.0f - lv) * hi;
    }
  }
}

// ---------------------------------------------------------------------------
extern "C" void kernel_launch(void* const* d_in, const int* in_sizes, int n_in,
                              void* d_out, int out_size, void* d_ws, size_t ws_size,
                              hipStream_t stream) {
  const float* x         = (const float*)d_in[0];
  const float* h         = (const float*)d_in[1];
  const float* c         = (const float*)d_in[2];
  const float* rm        = (const float*)d_in[3];
  const float* conv_w    = (const float*)d_in[4];
  const float* conv_b    = (const float*)d_in[5];
  const float* convc_w   = (const float*)d_in[6];
  const float* convc_b   = (const float*)d_in[7];
  const float* convssm_w = (const float*)d_in[8];
  const float* convssm_b = (const float*)d_in[9];
  const float* convfuse_w= (const float*)d_in[10];
  const float* convfuse_b= (const float*)d_in[11];

  // workspace layout (all offsets 256B aligned)
  const size_t o_wfrag = 0;                // 737280 B
  const size_t o_wfuse = 737280;           // 16384 B
  const size_t o_wssm  = 753664;           // 98304 B
  const size_t o_qavg  = 851968;           // 12288 B
  const size_t o_rmavg = 864256;           // 61440 B
  const size_t o_attn  = 925696;           // 1280 B
  const size_t o_mtcl  = 926976;           // 8388608 B
  const size_t o_mtup  = 9315584;          // 33554432 B
  const size_t o_xhcl  = 42870016;         // 67108864 B
  const size_t need    = 109978880;
  if (ws_size < need) return;

  char* ws = (char*)d_ws;
  _Float16* w_frag  = (_Float16*)(ws + o_wfrag);
  _Float16* wfuse   = (_Float16*)(ws + o_wfuse);
  _Float16* wssm    = (_Float16*)(ws + o_wssm);
  float*    q_avg   = (float*)(ws + o_qavg);
  float*    rm_avg  = (float*)(ws + o_rmavg);
  float*    attn    = (float*)(ws + o_attn);
  _Float16* mt_cl   = (_Float16*)(ws + o_mtcl);
  _Float16* mtup    = (_Float16*)(ws + o_mtup);
  _Float16* xh_cl   = (_Float16*)(ws + o_xhcl);

  float* hout = (float*)d_out;
  float* cout = (float*)d_out + (long)NB * HC * PIX;

  k0_weights<<<1664, 256, 0, stream>>>(conv_w, convfuse_w, convssm_w, w_frag, wfuse, wssm);
  kx_xh<<<NB * 32, 256, 0, stream>>>(x, h, xh_cl);
  kzero<<<12, 256, 0, stream>>>(q_avg);
  k1a_rmavg<<<NB * REC * MG, 256, 0, stream>>>(rm, rm_avg);
  k1b_qavg<<<NB * 16, 256, 0, stream>>>(c, convc_w, convc_b, q_avg);
  k1c_attn<<<NB, 32, 0, stream>>>(q_avg, rm_avg, attn);
  k2a_mt<<<NB * 64, 256, 0, stream>>>(rm, attn, mt_cl);
  k2b_ssm<<<NB * 4 * 16, 256, 0, stream>>>(mt_cl, wssm, convssm_b, mtup);
  k3_main<<<NB * 32, 320, 0, stream>>>(xh_cl, h, c, w_frag, conv_b, wfuse, convfuse_b,
                                       mtup, hout, cout);
}
